// BiLSTM_CRF_69544110457471
// MI455X (gfx1250) — matrix operand order repaired and checked by reading, not compiled, not run on hardware
//
#include <hip/hip_runtime.h>
#include <hip/hip_bf16.h>
#include <math.h>
#include <stdint.h>

// ---------------------------------------------------------------------------
// BiLSTM-CRF for MI455X (gfx1250, wave32, WMMA 16x16x32 bf16)
//   K1: embedding gather + input GEMM  -> gate pre-activations (bf16)
//   K2: persistent recurrent scan      -> hcat (bf16); WMMA per step with
//       async-to-LDS double-buffered gate tiles (ASYNCcnt) + L2 prefetch
//   K3: emission GEMM                  -> emis (f32, K padded to 16)
//   K4: Viterbi decode per batch elem  -> scores + paths
// ---------------------------------------------------------------------------

typedef __bf16 bf16_t;
typedef bf16_t v16bf __attribute__((ext_vector_type(16)));
typedef bf16_t v8bf  __attribute__((ext_vector_type(8)));
typedef float  v8f   __attribute__((ext_vector_type(8)));

#define T_LEN 512
#define B_SZ  128
#define E_DIM 100
#define KP    128   // E padded to multiple of 32 (WMMA K)
#define HD    64
#define G4    256   // 4*HD gates
#define KTAGS 12

__device__ __forceinline__ bf16_t f2bf(float x) { return (bf16_t)x; }
__device__ __forceinline__ float  sigf(float x) { return 1.0f / (1.0f + __expf(-x)); }

// Hardware tanh (V_TANH_F32 is a CDNA5 TRANS op) when the builtin exists.
#if defined(__has_builtin)
#  if __has_builtin(__builtin_amdgcn_tanhf)
#    define TANHF(x) __builtin_amdgcn_tanhf(x)
#  endif
#endif
#ifndef TANHF
#  define TANHF(x) tanhf(x)
#endif

// Load one WMMA bf16 operand fragment as two contiguous 8-half (16B) groups.
__device__ __forceinline__ v16bf ld_frag(const bf16_t* row, int k0, int k1) {
  v16bf r;
  ((v8bf*)&r)[0] = *(const v8bf*)(row + k0);
  ((v8bf*)&r)[1] = *(const v8bf*)(row + k1);
  return r;
}

// ---- CDNA5 async global->LDS copy (tracked by ASYNCcnt) -------------------
#define AS3 __attribute__((address_space(3)))
__device__ __forceinline__ unsigned lds_off(void* p) {
  // generic -> LDS addrspace cast, then ptrtoint gives the LDS byte offset
  return (unsigned)(unsigned long long)(AS3 void*)p;
}
__device__ __forceinline__ void async_ld_b128(unsigned ldsOff, const void* g) {
  asm volatile("global_load_async_to_lds_b128 %0, %1, off"
               :: "v"(ldsOff), "v"((unsigned long long)(uintptr_t)g)
               : "memory");
}
__device__ __forceinline__ void wait_async0() {
  asm volatile("s_wait_asynccnt 0x0" ::: "memory");
}

// ---------------------------------------------------------------------------
// K1: gates[dir][t][b][n] = emb[sent[b][t]] @ Wih^T + bih + bhh   (bf16 out)
// grid (1024, 2), block 256.  Per WG: 64 rows x 256 cols, K=128 (padded).
// ---------------------------------------------------------------------------
__global__ void gates_gemm_kernel(const int* __restrict__ sent,
                                  const float* __restrict__ emb,
                                  const float* __restrict__ Wih_f,
                                  const float* __restrict__ bih_f,
                                  const float* __restrict__ bhh_f,
                                  const float* __restrict__ Wih_b,
                                  const float* __restrict__ bih_b,
                                  const float* __restrict__ bhh_b,
                                  bf16_t* __restrict__ gates) {
  __shared__ __align__(16) bf16_t As[64][KP];
  __shared__ __align__(16) bf16_t Bs[G4][KP];
  const int dir  = blockIdx.y;
  const int tid  = threadIdx.x;
  const int lane = tid & 31, wv = tid >> 5;
  const float* Wih = dir ? Wih_b : Wih_f;
  const float* bih = dir ? bih_b : bih_f;
  const float* bhh = dir ? bhh_b : bhh_f;

  { // Stage B = Wih (row n holds K contiguous) as bf16, zero-padded to K=128.
    const float* src = Wih + (size_t)tid * E_DIM;
    for (int k = 0; k < KP; ++k)
      Bs[tid][k] = (k < E_DIM) ? f2bf(src[k]) : f2bf(0.0f);
  }
  // Stage A = gathered embedding rows (row = t*B + b).
  const int row0 = blockIdx.x * 64;
  {
    int r = tid >> 2, q = tid & 3;
    int gr = row0 + r;
    int t = gr >> 7, b = gr & 127;
    int tok = sent[b * T_LEN + t];
    const float* src = emb + (size_t)tok * E_DIM;
    for (int j = 0; j < 32; ++j) {
      int k = q * 32 + j;
      As[r][k] = (k < E_DIM) ? f2bf(src[k]) : f2bf(0.0f);
    }
  }
  __syncthreads();

  bf16_t* gout = gates + (size_t)dir * T_LEN * B_SZ * G4;
  const int abase = (lane & 16) ? 8 : 0;
  const int kbofs = (lane & 16) ? 16 : 0;
  const int mofs  = (lane & 16) ? 8 : 0;
  for (int i = 0; i < 8; ++i) {           // 8 tiles per wave (4 Mt x 16 Nt / 8 waves)
    int tt = wv * 8 + i;
    int mt = tt >> 4, nt = tt & 15;
    int n = nt * 16 + (lane & 15);
    float bias = bih[n] + bhh[n];
    v8f c;
    for (int r = 0; r < 8; ++r) c[r] = bias;
    const bf16_t* arow = &As[mt * 16 + (lane & 15)][0];
    for (int kk = 0; kk < 4; ++kk) {
      v16bf a   = ld_frag(arow,       kk * 32 + abase, kk * 32 + 16 + abase);
      v16bf bfr = ld_frag(&Bs[n][0],  kk * 32 + abase, kk * 32 + 16 + abase);
      c = __builtin_amdgcn_wmma_f32_16x16x32_bf16(false, a, false, bfr,
                                                  (short)0, c, false, false);
    }
    for (int r = 0; r < 8; ++r) {
      int gm = row0 + mt * 16 + r + mofs;
      int t = gm >> 7, b = gm & 127;
      gout[((size_t)t * B_SZ + b) * G4 + n] = f2bf(c[r]);
    }
  }
}

// ---------------------------------------------------------------------------
// K2: persistent recurrent scan.  grid 16 (dir*8 + batch-tile), block 256.
// Each WG owns 16 batch rows.  The 8KB gate tile for step s+1 is pulled into
// LDS with GLOBAL_LOAD_ASYNC_TO_LDS_B128 while step s computes
// h@Whh^T via WMMA; step s+2's tile is prefetched into L2.
// ---------------------------------------------------------------------------
__global__ void lstm_scan_kernel(const bf16_t* __restrict__ gates,
                                 const float* __restrict__ Whh_f,
                                 const float* __restrict__ Whh_b,
                                 const float* __restrict__ h0,
                                 const float* __restrict__ c0,
                                 bf16_t* __restrict__ hcat) {
  __shared__ __align__(16) bf16_t Wb[G4][HD];      // Whh as B-matrix (32 KB)
  __shared__ __align__(16) bf16_t hb[16][HD];      // recurrent h, bf16
  __shared__ __align__(16) bf16_t gbuf[2][16 * G4];// double-buffered gate tiles (16 KB)
  __shared__ float cs[16][HD];                     // cell state, f32
  __shared__ float gsm[16][G4];                    // per-step gate exchange (16 KB)
  const int dir     = blockIdx.x >> 3;
  const int rowbase = (blockIdx.x & 7) * 16;
  const int tid = threadIdx.x, lane = tid & 31, wv = tid >> 5;
  const float* Whh = dir ? Whh_b : Whh_f;

  const bf16_t* gdir = gates + (size_t)dir * T_LEN * B_SZ * G4;
  auto tile_ptr = [&](int ss) -> const bf16_t* {   // contiguous 8 KB per step
    int tt = dir ? (T_LEN - 1 - ss) : ss;
    return gdir + ((size_t)tt * B_SZ + rowbase) * G4;
  };
  const unsigned gOff = lds_off(&gbuf[0][0]);

  // Kick off async copy of step-0 gate tile while we stage Whh / h0 / c0.
  {
    const char* src = (const char*)tile_ptr(0);
    async_ld_b128(gOff + tid * 16,        src + tid * 16);
    async_ld_b128(gOff + 4096 + tid * 16, src + 4096 + tid * 16);
  }
  { // load Whh row per thread
    const float* src = Whh + (size_t)tid * HD;
    for (int k = 0; k < HD; ++k) Wb[tid][k] = f2bf(src[k]);
  }
  for (int j = 0; j < 4; ++j) {                    // init h, c
    int p = tid + 256 * j;
    int r = p >> 6, hd = p & 63;
    int b = rowbase + r;
    hb[r][hd] = f2bf(h0[((size_t)dir * B_SZ + b) * HD + hd]);
    cs[r][hd] =       c0[((size_t)dir * B_SZ + b) * HD + hd];
  }
  wait_async0();
  __syncthreads();

  const int mofs  = (lane & 16) ? 8 : 0;
  const int abase = (lane & 16) ? 8 : 0;
  const int kbofs = (lane & 16) ? 16 : 0;
  const int mloc  = lane & 15;

  for (int s = 0; s < T_LEN; ++s) {
    int t = dir ? (T_LEN - 1 - s) : s;
    // async stage next step's tile into the other buffer (overlaps compute)
    if (s + 1 < T_LEN) {
      const char* src = (const char*)tile_ptr(s + 1);
      unsigned dst = gOff + ((unsigned)((s + 1) & 1)) * 8192u;
      async_ld_b128(dst + tid * 16,        src + tid * 16);
      async_ld_b128(dst + 4096 + tid * 16, src + 4096 + tid * 16);
    }
    if (s + 2 < T_LEN)   // warm L2 two steps ahead (global_prefetch_b8)
      __builtin_prefetch((const char*)tile_ptr(s + 2) + tid * 32, 0, 2);

    const bf16_t* gcur = &gbuf[s & 1][0];
    for (int nti = 0; nti < 2; ++nti) {            // each wave owns 2 N-tiles
      int nt = wv * 2 + nti;
      int n = nt * 16 + (lane & 15);
      v8f c;
      for (int r = 0; r < 8; ++r)                  // C init = staged gates (LDS)
        c[r] = (float)gcur[(r + mofs) * G4 + n];
      for (int kk = 0; kk < 2; ++kk) {             // K = 64 = 2 x 32
        v16bf a   = ld_frag(&hb[mloc][0], kk * 32 + abase, kk * 32 + 16 + abase);
        v16bf bfr = ld_frag(&Wb[n][0],    kk * 32 + abase, kk * 32 + 16 + abase);
        c = __builtin_amdgcn_wmma_f32_16x16x32_bf16(false, a, false, bfr,
                                                    (short)0, c, false, false);
      }
      for (int r = 0; r < 8; ++r)
        gsm[r + mofs][n] = c[r];
    }
    __syncthreads();
    for (int j = 0; j < 4; ++j) {                  // LSTM cell (i,f,g,o order)
      int p = tid + 256 * j;
      int r = p >> 6, hd = p & 63;
      float gi = gsm[r][hd];
      float gf = gsm[r][64 + hd];
      float gg = gsm[r][128 + hd];
      float go = gsm[r][192 + hd];
      float cv = sigf(gf) * cs[r][hd] + sigf(gi) * TANHF(gg);
      float hv = sigf(go) * TANHF(cv);
      cs[r][hd] = cv;
      hb[r][hd] = f2bf(hv);
      int b = rowbase + r;
      hcat[((size_t)t * B_SZ + b) * 128 + dir * HD + hd] = f2bf(hv);
    }
    wait_async0();      // next buffer fully in LDS before anyone reads it
    __syncthreads();
  }
}

// ---------------------------------------------------------------------------
// K3: emis[b][t][k] = hcat[t][b][:] @ W_out^T + b_out  (K padded 12->16)
// grid 512, block 256.  Per WG: 128 rows x 16 cols, K=128.
// ---------------------------------------------------------------------------
__global__ void emis_kernel(const bf16_t* __restrict__ hcat,
                            const float* __restrict__ W_out,
                            const float* __restrict__ b_out,
                            float* __restrict__ emis) {
  __shared__ __align__(16) bf16_t As[128][128];
  __shared__ __align__(16) bf16_t Bs[16][128];
  const int tid = threadIdx.x, lane = tid & 31, wv = tid >> 5;
  const int row0 = blockIdx.x * 128;
  { // A: straight 32 KB copy of bf16 hcat rows (already row = t*B+b)
    const uint4* src = (const uint4*)(hcat + (size_t)row0 * 128);
    uint4* dst = (uint4*)&As[0][0];
    for (int j = 0; j < 8; ++j) dst[tid + 256 * j] = src[tid + 256 * j];
  }
  { // B: W_out rows, zero-padded to 16
    int n = tid >> 4, k0 = (tid & 15) * 8;
    for (int j = 0; j < 8; ++j) {
      int k = k0 + j;
      Bs[n][k] = (n < KTAGS) ? f2bf(W_out[n * 128 + k]) : f2bf(0.0f);
    }
  }
  __syncthreads();
  const int mofs  = (lane & 16) ? 8 : 0;
  const int abase = (lane & 16) ? 8 : 0;
  const int kbofs = (lane & 16) ? 16 : 0;
  int n = lane & 15;
  float bias = (n < KTAGS) ? b_out[n] : 0.0f;
  v8f c;
  for (int r = 0; r < 8; ++r) c[r] = bias;
  const bf16_t* arow = &As[wv * 16 + (lane & 15)][0];
  for (int kk = 0; kk < 4; ++kk) {
    v16bf a   = ld_frag(arow,      kk * 32 + abase, kk * 32 + 16 + abase);
    v16bf bfr = ld_frag(&Bs[n][0], kk * 32 + abase, kk * 32 + 16 + abase);
    c = __builtin_amdgcn_wmma_f32_16x16x32_bf16(false, a, false, bfr,
                                                (short)0, c, false, false);
  }
  for (int r = 0; r < 8; ++r) {
    int gm = row0 + wv * 16 + r + mofs;
    int t = gm >> 7, b = gm & 127;
    emis[((size_t)b * T_LEN + t) * 16 + n] = c[r];
  }
}

// ---------------------------------------------------------------------------
// K4: Viterbi. grid 128 (one wave per batch element), block 32.
// lane = to-tag; shfl-broadcast delta; backpointers in LDS; lane0 backtrace.
// out[0..127] = scores, out[128 + b*512 + t] = path tags (as float).
// ---------------------------------------------------------------------------
__global__ void viterbi_kernel(const float* __restrict__ emis,
                               const float* __restrict__ trans,
                               float* __restrict__ out) {
  __shared__ unsigned char bp[T_LEN][KTAGS];
  const int b = blockIdx.x;
  const int lane = threadIdx.x;
  float tc[KTAGS];
  for (int kf = 0; kf < KTAGS; ++kf)
    tc[kf] = (lane < KTAGS) ? trans[kf * KTAGS + lane] : 0.0f;
  const float* em = emis + (size_t)b * T_LEN * 16;
  float delta = (lane < KTAGS) ? em[lane] : -3.0e38f;
  for (int t = 1; t < T_LEN; ++t) {
    float best = -3.0e38f; int arg = 0;
    for (int kf = 0; kf < KTAGS; ++kf) {        // strict > keeps first argmax
      float v = __shfl(delta, kf, 32) + tc[kf];
      if (v > best) { best = v; arg = kf; }
    }
    if (lane < KTAGS) {
      bp[t][lane] = (unsigned char)arg;
      delta = em[t * 16 + lane] + best;
    }
  }
  __syncthreads();
  float bestd = -3.0e38f; int bt = 0;
  for (int k = 0; k < KTAGS; ++k) {
    float dv = __shfl(delta, k, 32);
    if (dv > bestd) { bestd = dv; bt = k; }
  }
  if (lane == 0) {
    out[b] = bestd;
    float* paths = out + B_SZ;
    int tag = bt;
    paths[(size_t)b * T_LEN + (T_LEN - 1)] = (float)tag;
    for (int t = T_LEN - 2; t >= 0; --t) {
      tag = bp[t + 1][tag];
      paths[(size_t)b * T_LEN + t] = (float)tag;
    }
  }
}

// ---------------------------------------------------------------------------
extern "C" void kernel_launch(void* const* d_in, const int* in_sizes, int n_in,
                              void* d_out, int out_size, void* d_ws, size_t ws_size,
                              hipStream_t stream) {
  (void)in_sizes; (void)n_in; (void)out_size; (void)ws_size;
  const int*   sent  = (const int*)d_in[0];
  const float* emb   = (const float*)d_in[1];
  const float* Wih_f = (const float*)d_in[2];
  const float* Whh_f = (const float*)d_in[3];
  const float* bih_f = (const float*)d_in[4];
  const float* bhh_f = (const float*)d_in[5];
  const float* Wih_b = (const float*)d_in[6];
  const float* Whh_b = (const float*)d_in[7];
  const float* bih_b = (const float*)d_in[8];
  const float* bhh_b = (const float*)d_in[9];
  const float* W_out = (const float*)d_in[10];
  const float* b_out = (const float*)d_in[11];
  const float* trans = (const float*)d_in[12];
  const float* h0    = (const float*)d_in[13];
  const float* c0    = (const float*)d_in[14];
  float* out = (float*)d_out;

  // workspace layout
  bf16_t* gates = (bf16_t*)d_ws;                         // 2*512*128*256 bf16 = 64 MB
  bf16_t* hcat  = gates + (size_t)2 * T_LEN * B_SZ * G4; // 512*128*128 bf16 = 16 MB
  float*  emis  = (float*)(hcat + (size_t)T_LEN * B_SZ * 128); // 128*512*16 f32 = 4 MB

  gates_gemm_kernel<<<dim3(1024, 2), 256, 0, stream>>>(
      sent, emb, Wih_f, bih_f, bhh_f, Wih_b, bih_b, bhh_b, gates);
  lstm_scan_kernel<<<16, 256, 0, stream>>>(gates, Whh_f, Whh_b, h0, c0, hcat);
  emis_kernel<<<512, 256, 0, stream>>>(hcat, W_out, b_out, emis);
  viterbi_kernel<<<128, 32, 0, stream>>>(emis, trans, out);
}
